// VersorRotorRNN_23502061043874
// MI455X (gfx1250) — matrix operand order repaired and checked by reading, not compiled, not run on hardware
//
#include <hip/hip_runtime.h>

// ---------------------------------------------------------------------------
// VersorRotorRNN for MI455X (gfx1250).
// Sequential scan, latency-bound: one persistent cooperative kernel,
// bf16 WMMA (v_wmma_f32_16x16x32_bf16) for all GEMMs, device-scope atomic
// grid barrier per timestep, Cayley-sparsity exploited on device.
// ---------------------------------------------------------------------------

typedef float  v8f   __attribute__((ext_vector_type(8)));
typedef __bf16 v16bf __attribute__((ext_vector_type(16)));
typedef __bf16 v8bf  __attribute__((ext_vector_type(8)));

#define B_SZ   256
#define S_LEN  1024
#define ND     30
#define HM     512     // H*M = 16*32
#define NBLK   32
#define WGS    256

// ---- workspace layout (bytes) ----
constexpr size_t OFF_WHB   = 0;         // 512*512 bf16 swizzled      = 524288
constexpr size_t OFF_WCB   = 524288;    // 32*512  bf16 swizzled      = 32768
constexpr size_t OFF_WOB   = 557056;    // 512*32  bf16 swizzled      = 32768
constexpr size_t OFF_BCOMB = 589824;    // 512 f32                    = 2048
constexpr size_t OFF_CNT   = 591872;    // barrier counter            = 256
constexpr size_t OFF_H0    = 592128;    // 256*512 bf16               = 262144
constexpr size_t OFF_H1    = 854272;    // 256*512 bf16               = 262144
constexpr size_t OFF_XBF   = 1116416;   // 256*1024*32 bf16           = 16777216
constexpr size_t WS_NEED   = OFF_XBF + 16777216;

// sign of e_m * e_n -> e_{m^n} in Cl(4,1): metric (+,+,+,+,-)
__device__ __forceinline__ float cayley_sign(int m, int n) {
  int par = 0;
  int t = m >> 1;
  while (t) { par ^= (__popc(t & n) & 1); t >>= 1; }
  par ^= ((m & n) >> 4) & 1;              // generator 4 squares to -1
  return par ? -1.0f : 1.0f;
}

__device__ __forceinline__ v8f wmma_bf16(v16bf a, v16bf b, v8f c) {
  return __builtin_amdgcn_wmma_f32_16x16x32_bf16(
      false, a, false, b, (short)0, c, false, false);
}

// A-matrix (16x32 bf16) fragment load from a row-major [*, stride] buffer.
// rowptr already points at (row, kb*32); koff = (lane&16)?8:0.
__device__ __forceinline__ v16bf load_a16(const __bf16* rowptr, int koff) {
  v8bf lo = *(const v8bf*)(rowptr + 2 * koff);
  v8bf hi = *(const v8bf*)(rowptr + 2 * koff + 8);
  v16bf a;
#pragma unroll
  for (int i = 0; i < 8; ++i) { a[i] = lo[i]; a[i + 8] = hi[i]; }
  return a;
}

// ---------------------------------------------------------------------------
// Prep kernels: build swizzled bf16 weight images on device.
// Swizzled B layout: elem[((blk)*32 + lane)*16 + j] = W[k, n]
//   k = kbase + (lane<16 ? j : 16+j), n = nb*16 + (lane&15)
// ---------------------------------------------------------------------------

__global__ void build_whb(const float* __restrict__ w_h, __bf16* __restrict__ WhB) {
  int idx  = blockIdx.x * 256 + threadIdx.x;        // < 262144
  int j    = idx & 15;
  int lane = (idx >> 4) & 31;
  int nb   = (idx >> 9) & 31;
  int kb   = idx >> 14;
  int r    = kb * 32 + ((lane & 16) ? 16 + j : j);  // row of Wh = (i,m)
  int n    = nb * 16 + (lane & 15);                 // col of Wh = (o,k)
  int i = r >> 5, m = r & 31;
  int o = n >> 5, kk = n & 31;
  int src = m ^ kk;
  float v = cayley_sign(m, src) * w_h[(o * 16 + i) * 32 + src];
  WhB[idx] = (__bf16)v;
}

__global__ void build_wcombb(const float* __restrict__ W_in,
                             const float* __restrict__ w_x,
                             __bf16* __restrict__ WcB) {
  int idx  = blockIdx.x * 256 + threadIdx.x;        // < 16384
  int j    = idx & 15;
  int lane = (idx >> 4) & 31;
  int nb   = (idx >> 9) & 31;
  int k    = (lane & 16) ? 16 + j : j;              // input dim d (pad 30,31)
  int n    = nb * 16 + (lane & 15);
  float v = 0.0f;
  if (k < ND) {
    int o = n >> 5, kk = n & 31;
    for (int i = 0; i < 16; ++i) {
      for (int m = 0; m < 32; ++m) {
        int src = m ^ kk;
        v += W_in[k * HM + i * 32 + m] * cayley_sign(m, src) *
             w_x[(o * 16 + i) * 32 + src];
      }
    }
  }
  WcB[idx] = (__bf16)v;
}

__global__ void build_bcomb(const float* __restrict__ b_in,
                            const float* __restrict__ w_x,
                            float* __restrict__ bcomb) {
  int n = blockIdx.x * 256 + threadIdx.x;           // < 512
  if (n >= HM) return;
  int o = n >> 5, kk = n & 31;
  float v = 0.0f;
  for (int i = 0; i < 16; ++i)
    for (int m = 0; m < 32; ++m) {
      int src = m ^ kk;
      v += b_in[i * 32 + m] * cayley_sign(m, src) * w_x[(o * 16 + i) * 32 + src];
    }
  bcomb[n] = v;
}

__global__ void build_woutb(const float* __restrict__ W_out, __bf16* __restrict__ WoB) {
  int idx   = blockIdx.x * 256 + threadIdx.x;       // < 16384
  int j     = idx & 15;
  int lane  = (idx >> 4) & 31;
  int nbAll = idx >> 9;
  int nb    = nbAll & 1;
  int kb    = nbAll >> 1;
  int k   = kb * 32 + ((lane & 16) ? 16 + j : j);
  int col = nb * 16 + (lane & 15);
  float v = (col < ND) ? W_out[k * ND + col] : 0.0f;
  WoB[idx] = (__bf16)v;
}

__global__ void build_xbf(const float* __restrict__ x, __bf16* __restrict__ Xbf) {
  size_t idx = (size_t)blockIdx.x * 256 + threadIdx.x;  // < 256*1024*32
  int k = (int)(idx & 31);
  size_t bt = idx >> 5;
  float v = (k < ND) ? x[bt * ND + k] : 0.0f;
  Xbf[idx] = (__bf16)v;
}

// ---------------------------------------------------------------------------
// Persistent recurrent kernel. 32 blocks x 256 threads (8 wave32 each).
// wave = global wave id (0..255): ch = wave>>4 (multivector channel),
// mt = wave&15 (batch tile). Each wave owns the 16x32 tile
// rows [mt*16, mt*16+16), cols [ch*32, ch*32+32) of h.
// ---------------------------------------------------------------------------

__global__ __launch_bounds__(WGS) void rnn_persistent(
    const float* __restrict__ x,        // [B,S,30]
    const float* __restrict__ b_out,    // [30]
    const __bf16* __restrict__ WhB,
    const __bf16* __restrict__ WcB,
    const __bf16* __restrict__ WoB,
    const float* __restrict__ bcomb,
    const __bf16* __restrict__ Xbf,     // [B,S,32]
    __bf16* __restrict__ hbuf0,
    __bf16* __restrict__ hbuf1,
    unsigned* __restrict__ bar_cnt,
    float* __restrict__ out)
{
  __shared__ __attribute__((aligned(32))) __bf16 ldsWh[16 * 2 * 32 * 16]; // 32KB
  __shared__ __attribute__((aligned(32))) __bf16 ldsWc[2 * 32 * 16];      // 2KB
  __shared__ __attribute__((aligned(16))) float  ldsRed[8 * 256];         // 8KB

  const int tid  = threadIdx.x;
  const int g    = blockIdx.x;
  const int wave = (g * WGS + tid) >> 5;   // 0..255
  const int lane = tid & 31;
  const int ch   = wave >> 4;              // 0..15 (uniform per WG: ch == g>>1)
  const int mt   = wave & 15;
  const int b0   = mt * 16;
  const int lc   = lane & 15;
  const int koff = (lane & 16) ? 8 : 0;    // A-fragment K sub-offset

  // stage this WG's Wh column-slice (channel ch -> nb blocks 2ch, 2ch+1) + Wcomb slice
  {
    const int wgch = g >> 1;
    for (int i = tid; i < 16 * 1024; i += WGS) {
      int kb = i >> 10, rem = i & 1023;
      ldsWh[i] = WhB[((size_t)(kb * 32 + wgch * 2) * 32) * 16 + rem];
    }
    for (int i = tid; i < 1024; i += WGS)
      ldsWc[i] = WcB[((size_t)(wgch * 2) * 32) * 16 + i];
  }
  __syncthreads();

  const float bias0 = bcomb[ch * 32 + lc];
  const float bias1 = bcomb[ch * 32 + 16 + lc];

  const __bf16* hcur = hbuf0;
  __bf16*       hnxt = hbuf1;
  unsigned bar_target = 0;

  const int omt = g & 15, oct = g >> 4;    // this WG's output-projection tile
  const int ob0 = omt * 16;
  const int wIn = (tid >> 5) & 7;          // wave index inside WG

  for (int t = 0; t < S_LEN; ++t) {
    v8f acc0, acc1;
#pragma unroll
    for (int r = 0; r < 8; ++r) { acc0[r] = bias0; acc1[r] = bias1; }

    // x-side: x_t @ Wcomb (K=32, one WMMA per n-block)
    {
      const __bf16* rp = Xbf + ((size_t)(b0 + lc) * S_LEN + t) * 32;
      v16bf a  = load_a16(rp, koff);
      v16bf bx0 = *(const v16bf*)&ldsWc[(0 * 32 + lane) * 16];
      v16bf bx1 = *(const v16bf*)&ldsWc[(1 * 32 + lane) * 16];
      acc0 = wmma_bf16(a, bx0, acc0);
      acc1 = wmma_bf16(a, bx1, acc1);
    }
    // recurrent: h @ Wh, K = 512 in 16 blocks of 32
#pragma unroll 4
    for (int kb = 0; kb < 16; ++kb) {
      const __bf16* rp = hcur + (size_t)(b0 + lc) * HM + kb * 32;
      v16bf a  = load_a16(rp, koff);
      v16bf bb0 = *(const v16bf*)&ldsWh[((kb * 2 + 0) * 32 + lane) * 16];
      v16bf bb1 = *(const v16bf*)&ldsWh[((kb * 2 + 1) * 32 + lane) * 16];
      acc0 = wmma_bf16(a, bb0, acc0);
      acc1 = wmma_bf16(a, bb1, acc1);
    }

    // residual + per-channel L2 normalization + store h_new (bf16, C-layout rows)
#pragma unroll
    for (int r = 0; r < 8; ++r) {
      int row  = b0 + r + ((lane & 16) ? 8 : 0);
      int base = row * HM + ch * 32;
      float v0 = acc0[r] + (float)hcur[base + lc];
      float v1 = acc1[r] + (float)hcur[base + 16 + lc];
      float ss = v0 * v0 + v1 * v1;
      ss += __shfl_xor(ss, 1, 32);
      ss += __shfl_xor(ss, 2, 32);
      ss += __shfl_xor(ss, 4, 32);
      ss += __shfl_xor(ss, 8, 32);
      float scale = 1.0f / (sqrtf(ss) + 1e-6f);
      hnxt[base + lc]      = (__bf16)(v0 * scale);
      hnxt[base + 16 + lc] = (__bf16)(v1 * scale);
    }

    // ---- device-wide barrier (release -> count -> acquire) ----
    __builtin_amdgcn_fence(__ATOMIC_RELEASE, "agent");
    __syncthreads();
    bar_target += NBLK;
    if (tid == 0) {
      __hip_atomic_fetch_add(bar_cnt, 1u, __ATOMIC_RELEASE, __HIP_MEMORY_SCOPE_AGENT);
      while (__hip_atomic_load(bar_cnt, __ATOMIC_RELAXED, __HIP_MEMORY_SCOPE_AGENT)
             < bar_target) {
        __builtin_amdgcn_s_sleep(1);
      }
    }
    __syncthreads();
    __builtin_amdgcn_fence(__ATOMIC_ACQUIRE, "agent");

    // ---- output projection: out_t = x_t + h_new @ W_out + b_out ----
    // WG computes one 16x16 tile; K split over 8 waves, LDS reduction.
    {
      v8f oacc;
#pragma unroll
      for (int r = 0; r < 8; ++r) oacc[r] = 0.0f;
#pragma unroll
      for (int kk = 0; kk < 2; ++kk) {
        int kb = wIn * 2 + kk;
        const __bf16* rp = hnxt + (size_t)(ob0 + lc) * HM + kb * 32;
        v16bf a  = load_a16(rp, koff);
        v16bf bo = *(const v16bf*)&WoB[((size_t)(kb * 2 + oct) * 32 + lane) * 16];
        oacc = wmma_bf16(a, bo, oacc);
      }
#pragma unroll
      for (int r = 0; r < 8; ++r) ldsRed[wIn * 256 + r * 32 + lane] = oacc[r];
      __syncthreads();
      {
        float s = 0.0f;
#pragma unroll
        for (int w2 = 0; w2 < 8; ++w2) s += ldsRed[w2 * 256 + tid];
        int rl = tid & 31, rr = tid >> 5;
        int M   = rr + ((rl & 16) ? 8 : 0);
        int col = oct * 16 + (rl & 15);
        if (col < ND) {
          size_t addr = ((size_t)(ob0 + M) * S_LEN + t) * ND + col;
          out[addr] = x[addr] + b_out[col] + s;
        }
      }
      __syncthreads();   // protect ldsRed before next step
    }

    // swap double buffers
    const __bf16* tmp = hcur; hcur = hnxt; hnxt = (__bf16*)tmp;
  }
}

// ---------------------------------------------------------------------------
extern "C" void kernel_launch(void* const* d_in, const int* in_sizes, int n_in,
                              void* d_out, int out_size, void* d_ws, size_t ws_size,
                              hipStream_t stream) {
  (void)in_sizes; (void)n_in; (void)out_size;
  if (ws_size < WS_NEED) return;

  const float* x     = (const float*)d_in[0];
  const float* W_in  = (const float*)d_in[1];
  const float* b_in  = (const float*)d_in[2];
  const float* w_h   = (const float*)d_in[3];
  const float* w_x   = (const float*)d_in[4];
  const float* W_out = (const float*)d_in[5];
  const float* b_out = (const float*)d_in[6];

  char* ws = (char*)d_ws;
  __bf16*   WhB   = (__bf16*)(ws + OFF_WHB);
  __bf16*   WcB   = (__bf16*)(ws + OFF_WCB);
  __bf16*   WoB   = (__bf16*)(ws + OFF_WOB);
  float*    bcomb = (float*)(ws + OFF_BCOMB);
  unsigned* cnt   = (unsigned*)(ws + OFF_CNT);
  __bf16*   h0    = (__bf16*)(ws + OFF_H0);
  __bf16*   h1    = (__bf16*)(ws + OFF_H1);
  __bf16*   Xbf   = (__bf16*)(ws + OFF_XBF);

  hipMemsetAsync(cnt, 0, 256, stream);
  hipMemsetAsync(h0, 0, (size_t)B_SZ * HM * 2, stream);   // h_0 = 0

  build_whb   <<<1024,  256, 0, stream>>>(w_h, WhB);
  build_wcombb<<<64,    256, 0, stream>>>(W_in, w_x, WcB);
  build_bcomb <<<2,     256, 0, stream>>>(b_in, w_x, bcomb);
  build_woutb <<<64,    256, 0, stream>>>(W_out, WoB);
  build_xbf   <<<32768, 256, 0, stream>>>(x, Xbf);

  rnn_persistent<<<NBLK, WGS, 0, stream>>>(
      x, b_out, WhB, WcB, WoB, bcomb, Xbf, h0, h1, cnt, (float*)d_out);
}
